// LTFGW_semirelaxed_37254546326111
// MI455X (gfx1250) — matrix operand order repaired and checked by reading, not compiled, not run on hardware
//
#include <hip/hip_runtime.h>
#include <hip/hip_bf16.h>
#include <math.h>

#define NN    6000
#define DEG   16
#define MM    17          // ego-net size (self + DEG)
#define NTMP  16          // templates
#define MTD   10          // template size
#define NF    128         // features
#define NCOL  (NTMP*MTD)  // 160
#define NITER 10
#define WPR   ((NN + 31) / 32)   // adjacency bitset words per row
#define REGU  0.1f

typedef _Float16 v16h __attribute__((ext_vector_type(16)));
typedef float    v8f  __attribute__((ext_vector_type(8)));

// ---------------- adjacency bitset ----------------
__global__ void zero_adj(unsigned* adj, int nwords) {
  int i = blockIdx.x * blockDim.x + threadIdx.x;
  if (i < nwords) adj[i] = 0u;
}

__global__ void build_adj(const int* __restrict__ ei, unsigned* __restrict__ adj) {
  int e = blockIdx.x * blockDim.x + threadIdx.x;
  if (e >= NN * DEG) return;
  int u = ei[e];            // src row of edge_index
  int v = ei[NN * DEG + e]; // dst row of edge_index
  atomicOr(&adj[(size_t)u * WPR + (v >> 5)], 1u << (v & 31));
  atomicOr(&adj[(size_t)v * WPR + (u >> 5)], 1u << (u & 31)); // max(A, A^T)
}

// ---------------- main kernel: one block per node ----------------
struct Smem {
  _Float16 F1[MM][NF];        // 4352 B  (ego-net features, f16 for WMMA)
  _Float16 F2[NCOL][NF];      // 40960 B (all template features, f16)
  float    Dot[MM][NCOL];     // 10880 B (F1 . F2^T, f32 accum)
  float    C2[NTMP][MTD][MTD];// 6400 B
  float    N1[MM];            // ||F1 row||^2
  float    N2[NCOL];          // ||F2 row||^2
  unsigned C1[MM];            // ego-net adjacency row bitmasks
  int      Nbr[MM];
}; // ~63.4 KB

__global__ __launch_bounds__(512)
void srfgw_kernel(const float* __restrict__ x,
                  const int*   __restrict__ ei,
                  const float* __restrict__ tmpl,
                  const float* __restrict__ tfeat,
                  const float* __restrict__ alpha0,
                  const unsigned* __restrict__ adj,
                  float* __restrict__ out) {
  __shared__ Smem sm;
  const int tid  = threadIdx.x;
  const int n    = blockIdx.x;
  const int lane = tid & 31;
  const int wave = tid >> 5;   // 16 waves

  // ---------- setup ----------
  for (int i = tid; i < MM; i += 512) sm.C1[i] = 0u;
  if (tid < MM) {
    int nb = (tid == 0) ? n : ei[NN * DEG + n * DEG + tid - 1];
    sm.Nbr[tid] = nb;
    // warm the scattered 512B feature rows (global_prefetch_b8)
    const float* row = x + (size_t)nb * NF;
    __builtin_prefetch(row, 0, 3);
    __builtin_prefetch(row + 32, 0, 3);
    __builtin_prefetch(row + 64, 0, 3);
    __builtin_prefetch(row + 96, 0, 3);
  }
  __syncthreads();

  for (int i = tid; i < MM * NF; i += 512) {
    int m = i / NF, f = i % NF;
    sm.F1[m][f] = (_Float16)x[(size_t)sm.Nbr[m] * NF + f];
  }
  for (int i = tid; i < NCOL * NF; i += 512) {
    int c = i / NF, f = i % NF;
    sm.F2[c][f] = (_Float16)tfeat[(size_t)c * NF + f];
  }
  for (int i = tid; i < NTMP * MTD * MTD; i += 512)
    ((float*)sm.C2)[i] = tmpl[i];
  // deterministic squared norms from the original f32 data
  if (tid < MM) {
    float s = 0.f;
    const float* row = x + (size_t)sm.Nbr[tid] * NF;
    for (int f = 0; f < NF; ++f) { float v = row[f]; s = fmaf(v, v, s); }
    sm.N1[tid] = s;
  }
  if (tid < NCOL) {
    float s = 0.f;
    const float* row = tfeat + (size_t)tid * NF;
    for (int f = 0; f < NF; ++f) { float v = row[f]; s = fmaf(v, v, s); }
    sm.N2[tid] = s;
  }
  // C1 row bitmasks from the global adjacency bitset (binary => C1^2 == C1)
  for (int i = tid; i < MM * MM; i += 512) {
    int a = i / MM, b = i % MM;
    int u = sm.Nbr[a], v = sm.Nbr[b];
    unsigned bit = (adj[(size_t)u * WPR + (v >> 5)] >> (v & 31)) & 1u;
    if (bit) atomicOr(&sm.C1[a], 1u << b);
  }
  __syncthreads();

  // ---------- WMMA: Dot = F1 (17x128) @ F2^T (128x160) ----------
  // 2 M-tiles x 10 N-tiles, K=128 -> 4 x v_wmma_f32_16x16x32_f16 per tile.
  // Rows 17..31 of A are CLAMPED (not zeroed): garbage rows of D are simply
  // never stored, so the loads stay unconditional (no EXEC divergence).
  for (int tile = wave; tile < 20; tile += 16) {
    const int mtile = tile / 10, ntile = tile % 10;
    const int mrow0 = mtile * 16 + (lane & 15);
    const int mrow  = (mrow0 < MM) ? mrow0 : (MM - 1);   // clamp
    const int ncol  = ntile * 16 + (lane & 15);          // always < 160
    const int hiA   = (lane & 16) ? 8 : 0;   // A layout: lanes 16-31 hold K+8 group
    const int hiB   = (lane & 16) ? 8 : 0;   // B layout: lanes 16-31 hold K+8 group

    // preload all 4 K-step fragments, then issue WMMAs back-to-back
    v16h af[4], bf[4];
    #pragma unroll
    for (int kk4 = 0; kk4 < 4; ++kk4) {
      const int kk = kk4 * 32;
      #pragma unroll
      for (int i = 0; i < 16; ++i) {
        int j = i >> 1, s = i & 1;
        int kl = ((j >> 2) << 4) + hiA + ((j & 3) << 1) + s; // 16-bit A VGPR layout
        af[kk4][i] = sm.F1[mrow][kk + kl];
      }
      #pragma unroll
      for (int i = 0; i < 16; ++i)
        bf[kk4][i] = sm.F2[ncol][kk + hiB + (i & 7) + ((i >> 3) << 4)];
    }
    v8f acc = {};
    #pragma unroll
    for (int kk4 = 0; kk4 < 4; ++kk4)
      acc = __builtin_amdgcn_wmma_f32_16x16x32_f16(false, af[kk4], false, bf[kk4],
                                                   (short)0, acc, false, false);

    // C/D layout: VGPR r -> M = r (lanes 0-15) / 8+r (lanes 16-31), N = lane&15
    #pragma unroll
    for (int r = 0; r < 8; ++r) {
      int m = mtile * 16 + ((lane & 16) ? 8 : 0) + r;
      if (m < MM) sm.Dot[m][ntile * 16 + (lane & 15)] = acc[r];
    }
  }
  __syncthreads();

  // ---------- mirror descent: wave = template, lane = ego-net row ----------
  const int  k   = wave;
  const int  a   = lane;
  const bool act = a < MM;
  const float aval  = alpha0[0];
  const float alpha = 1.f / (1.f + __expf(-aval));
  const float g2  = 2.f * alpha / REGU;         // scale of gw_grad / REG
  const float cmc = (1.f - alpha) / REGU;       // scale of Mcost / REG
  const unsigned mask = act ? sm.C1[a] : 0u;
  const float c1p = (float)__popc(mask) * (1.0f / 17.0f); // sum_b C1^2 * p
  const float pv  = 1.0f / 17.0f;

  float lt[MTD], mc[MTD];
  #pragma unroll
  for (int t = 0; t < MTD; ++t) {
    float dot = act ? sm.Dot[a][k * MTD + t] : 0.f;
    float m2  = act ? (sm.N1[a] + sm.N2[k * MTD + t] - 2.f * dot) : 0.f;
    mc[t] = cmc * m2;
    lt[t] = -5.13579844f;   // log(1/17) - log(10)
  }

  float q[MTD];
  for (int iter = 0; iter <= NITER; ++iter) {
    // lane-local row softmax -> T = p * softmax(lt)
    float mx = lt[0];
    #pragma unroll
    for (int t = 1; t < MTD; ++t) mx = fmaxf(mx, lt[t]);
    float Tt[MTD]; float ssum = 0.f;
    #pragma unroll
    for (int t = 0; t < MTD; ++t) { Tt[t] = __expf(lt[t] - mx); ssum += Tt[t]; }
    float inv = pv * __builtin_amdgcn_rcpf(ssum);
    #pragma unroll
    for (int t = 0; t < MTD; ++t) Tt[t] = act ? Tt[t] * inv : 0.f;
    // q[t] = column sums over the 17 rows (butterfly; inactive lanes give 0)
    #pragma unroll
    for (int t = 0; t < MTD; ++t) {
      float v = Tt[t];
      for (int off = 16; off >= 1; off >>= 1) v += __shfl_xor(v, off, 32);
      q[t] = v;
    }
    if (iter == NITER) break;
    // constC2[j] = sum_b q[b] * C2[k][j][b]^2
    float cc[MTD];
    #pragma unroll
    for (int j = 0; j < MTD; ++j) {
      float sj = 0.f;
      #pragma unroll
      for (int b = 0; b < MTD; ++b) { float c2 = sm.C2[k][j][b]; sj = fmaf(q[b], c2 * c2, sj); }
      cc[j] = sj;
    }
    // Y[a][t] = sum_b C1[a][b] * T[b][t]  (uniform-lane broadcast + mask-gated FMA)
    float Y[MTD];
    #pragma unroll
    for (int t = 0; t < MTD; ++t) Y[t] = 0.f;
    for (int b = 0; b < MM; ++b) {
      float bit = (float)((mask >> b) & 1u);
      #pragma unroll
      for (int t = 0; t < MTD; ++t) {
        float tb = __shfl(Tt[t], b, 32);
        Y[t] = fmaf(bit, tb, Y[t]);
      }
    }
    // Y2[a][s] = sum_t Y[t] * C2[k][s][t]; lt -= (alpha*gw + (1-alpha)*Mcost)/REG
    #pragma unroll
    for (int s2 = 0; s2 < MTD; ++s2) {
      float y2 = 0.f;
      #pragma unroll
      for (int t = 0; t < MTD; ++t) y2 = fmaf(Y[t], sm.C2[k][s2][t], y2);
      lt[s2] -= g2 * (c1p + cc[s2] - 2.f * y2) + mc[s2];
    }
  }

  if (lane == 0) {
    #pragma unroll
    for (int t = 0; t < MTD; ++t)
      out[(size_t)n * NCOL + k * MTD + t] = q[t];
  }
}

extern "C" void kernel_launch(void* const* d_in, const int* in_sizes, int n_in,
                              void* d_out, int out_size, void* d_ws, size_t ws_size,
                              hipStream_t stream) {
  (void)in_sizes; (void)n_in; (void)out_size; (void)ws_size;
  const float* x     = (const float*)d_in[0];
  const int*   ei    = (const int*)d_in[1];
  const float* tmpl  = (const float*)d_in[2];
  const float* tfeat = (const float*)d_in[3];
  const float* a0    = (const float*)d_in[4];
  float*       out   = (float*)d_out;
  unsigned*    adj   = (unsigned*)d_ws;   // 6000*188*4 = 4.5 MB bitset (fits L2)

  const int nwords = NN * WPR;
  zero_adj<<<dim3((nwords + 255) / 256), dim3(256), 0, stream>>>(adj, nwords);
  build_adj<<<dim3((NN * DEG + 255) / 256), dim3(256), 0, stream>>>(ei, adj);
  srfgw_kernel<<<dim3(NN), dim3(512), 0, stream>>>(x, ei, tmpl, tfeat, a0, adj, out);
}
